// SGCrossAttention_88407606821435
// MI455X (gfx1250) — matrix operand order repaired and checked by reading, not compiled, not run on hardware
//
#include <hip/hip_runtime.h>
#include <hip/hip_bf16.h>

// ---------------------------------------------------------------------------
// Problem constants (from reference setup_inputs)
// B=8, T=2, BT=16, M=128, C=1024, CTX=768, H=16, D=64, ph=pw=48, N=2304
// ---------------------------------------------------------------------------
#define BT_  16
#define B_   8
#define T_   2
#define M_   128
#define C_   1024
#define CTX_ 768
#define H_   16
#define D_   64
#define N_   2304
#define HD_  1024

// LDS tile row pitch in halves (80 bytes: 16B-aligned, conflict-friendly)
#define LP_  40

#if defined(__AMDGCN__) && __has_builtin(__builtin_amdgcn_global_load_async_to_lds_b128)
#define HAVE_ASYNC 1
#else
#define HAVE_ASYNC 0
#endif

typedef __attribute__((ext_vector_type(16))) __bf16 v16bf;
typedef __attribute__((ext_vector_type(8)))  float  v8f;

struct B32x2 { uint4 lo; uint4 hi; };

static __device__ __forceinline__ v16bf make_frag(uint4 lo, uint4 hi) {
    B32x2 t{lo, hi};
    return __builtin_bit_cast(v16bf, t);
}

static __device__ __forceinline__ v8f wmma_bf16(v16bf a, v16bf b, v8f c) {
    return __builtin_amdgcn_wmma_f32_16x16x32_bf16(
        /*neg_a=*/false, a, /*neg_b=*/false, b,
        /*c_mod=*/(short)0, c, /*reuse_a=*/false, /*reuse_b=*/false);
}

static __device__ __forceinline__ unsigned short f2bf(float f) {
    unsigned u = __float_as_uint(f);
    unsigned r = u + 0x7FFFu + ((u >> 16) & 1u);  // round-to-nearest-even
    return (unsigned short)(r >> 16);
}

#if HAVE_ASYNC
typedef int v4i_gcc __attribute__((vector_size(16)));
typedef __attribute__((address_space(1))) v4i_gcc* g_v4i_ptr;
typedef __attribute__((address_space(3))) v4i_gcc* l_v4i_ptr;

static __device__ __forceinline__ void async_ld16(const unsigned short* g,
                                                  unsigned short* l) {
    __builtin_amdgcn_global_load_async_to_lds_b128(
        (g_v4i_ptr)(const_cast<unsigned short*>(g)),
        (l_v4i_ptr)l, 0, 0);
}
static __device__ __forceinline__ void wait_async4() {
#if __has_builtin(__builtin_amdgcn_s_wait_asynccnt)
    __builtin_amdgcn_s_wait_asynccnt(4);
#else
    asm volatile("s_wait_asynccnt 0x4" ::: "memory");
#endif
}
static __device__ __forceinline__ void wait_async0() {
#if __has_builtin(__builtin_amdgcn_s_wait_asynccnt)
    __builtin_amdgcn_s_wait_asynccnt(0);
#else
    asm volatile("s_wait_asynccnt 0x0" ::: "memory");
#endif
}
#endif

// ---------------------------------------------------------------------------
// Weight transpose + fp32 -> bf16 :  out[n*K + k] = in[k*Nn + n]
// ---------------------------------------------------------------------------
__global__ __launch_bounds__(256) void wtrans_kernel(
    const float* __restrict__ in, unsigned short* __restrict__ out,
    int K, int Nn) {
    int idx = blockIdx.x * 256 + threadIdx.x;
    if (idx >= K * Nn) return;
    int k = idx / Nn, n = idx % Nn;
    out[(size_t)n * K + k] = f2bf(in[idx]);
}

// ---------------------------------------------------------------------------
// Per-(bt, m) projection precompute: (u, v, z^2, flag)
// flag: -1 = masked(-inf), 0 = behind(bias 0), 1 = normal
// Rigid inverse: w2c = [R^T | -R^T t]
// ---------------------------------------------------------------------------
__global__ __launch_bounds__(256) void uvzf_kernel(
    const float* __restrict__ pos, const float* __restrict__ c2w,
    const float* __restrict__ intr, const unsigned char* __restrict__ mask,
    float4* __restrict__ out) {
    int idx = blockIdx.x * 256 + threadIdx.x;
    if (idx >= BT_ * M_) return;
    int bt = idx >> 7, m = idx & (M_ - 1);
    int b = bt >> 1;
    const float* Mx = c2w + (size_t)bt * 16;
    const float* p  = pos + (size_t)(b * M_ + m) * 3;
    float d0 = p[0] - Mx[3], d1 = p[1] - Mx[7], d2 = p[2] - Mx[11];
    float pcx = Mx[0] * d0 + Mx[4] * d1 + Mx[8]  * d2;
    float pcy = Mx[1] * d0 + Mx[5] * d1 + Mx[9]  * d2;
    float pcz = Mx[2] * d0 + Mx[6] * d1 + Mx[10] * d2;
    float z = fmaxf(pcz, 1e-4f);
    const float* I = intr + (size_t)b * 9;
    float u = I[0] * (pcx / z) + I[2];
    float v = I[4] * (pcy / z) + I[5];
    float w = mask[b * M_ + m] ? ((pcz <= 0.f) ? 0.f : 1.f) : -1.f;
    out[idx] = make_float4(u, v, z * z, w);
}

// ---------------------------------------------------------------------------
// Row LayerNorm, fp32 in -> bf16 out
// ---------------------------------------------------------------------------
__global__ __launch_bounds__(256) void ln_bf16_kernel(
    const float* __restrict__ in, const float* __restrict__ g,
    const float* __restrict__ bta, unsigned short* __restrict__ out, int C) {
    int row = blockIdx.x;
    int tid = threadIdx.x;
    const float* x = in + (size_t)row * C;
    float s1 = 0.f, s2 = 0.f;
    for (int i = tid; i < C; i += 256) { float v = x[i]; s1 += v; s2 += v * v; }
    __shared__ float r1[256], r2[256];
    r1[tid] = s1; r2[tid] = s2;
    __syncthreads();
    for (int o = 128; o > 0; o >>= 1) {
        if (tid < o) { r1[tid] += r1[tid + o]; r2[tid] += r2[tid + o]; }
        __syncthreads();
    }
    float mu = r1[0] / C;
    float var = r2[0] / C - mu * mu;
    float rstd = rsqrtf(var + 1e-5f);
    for (int i = tid; i < C; i += 256)
        out[(size_t)row * C + i] = f2bf((x[i] - mu) * rstd * g[i] + bta[i]);
}

// ---------------------------------------------------------------------------
// Tiled bf16 WMMA GEMM:  out = A(rows x K) @ Bt^T  (Bt stored [n][k], bf16)
// Block = 256 threads (8 waves). Block tile 128(M) x 128(N), K-chunks of 32.
// Double-buffered LDS staging; async (GLOBAL_LOAD_ASYNC_TO_LDS_B128 +
// s_wait_asynccnt) when the toolchain exposes the builtin, sync fallback else.
// Wave w owns rows [w*16, w*16+16) x all 128 cols (8 accumulators, A reused).
// MODE 0: bf16 row-major out
// MODE 1: bf16 out scattered to v^T layout [b][h][d][m]  (m packed, 16B stores)
// MODE 2: f32 out = xres + gate * acc
// ---------------------------------------------------------------------------
template <int MODE>
__global__ __launch_bounds__(256) void gemm_bf16_kernel(
    const unsigned short* __restrict__ A, const unsigned short* __restrict__ Bt,
    void* __restrict__ outp, const float* __restrict__ xres,
    const float* __restrict__ gatep, int K, int Nn,
    long long aBatch, long long oBatch) {
    __shared__ unsigned short lA[2][128 * LP_];
    __shared__ unsigned short lB[2][128 * LP_];
    const int tid = threadIdx.x;
    const int w = tid >> 5, lane = tid & 31;
    const int nl = lane & 15, s = lane >> 4;
    const int mbase = blockIdx.y * 128, nbase = blockIdx.x * 128;
    const int batch = blockIdx.z;
    const unsigned short* Ab = A + (size_t)batch * aBatch;

    const int row  = tid >> 1, half = tid & 1;       // staging assignment
    const int nk = K >> 5;

    v8f acc[8] = {};

#if HAVE_ASYNC
    {   // prefetch chunk 0 into buffer 0: 4 async b128 per thread
        const unsigned short* ga = Ab + (size_t)(mbase + row) * K + half * 16;
        const unsigned short* gb = Bt + (size_t)(nbase + row) * K + half * 16;
        unsigned short* la = &lA[0][row * LP_ + half * 16];
        unsigned short* lb = &lB[0][row * LP_ + half * 16];
        async_ld16(ga, la);     async_ld16(ga + 8, la + 8);
        async_ld16(gb, lb);     async_ld16(gb + 8, lb + 8);
    }
#endif

    for (int i = 0; i < nk; i++) {
        const int kb = i << 5;
        const int cur = i & 1;
#if HAVE_ASYNC
        if (i + 1 < nk) {   // prefetch next chunk into other buffer
            const int kn = kb + 32;
            const unsigned short* ga = Ab + (size_t)(mbase + row) * K + kn + half * 16;
            const unsigned short* gb = Bt + (size_t)(nbase + row) * K + kn + half * 16;
            unsigned short* la = &lA[cur ^ 1][row * LP_ + half * 16];
            unsigned short* lb = &lB[cur ^ 1][row * LP_ + half * 16];
            async_ld16(ga, la);     async_ld16(ga + 8, la + 8);
            async_ld16(gb, lb);     async_ld16(gb + 8, lb + 8);
            wait_async4();          // current buffer's 4 ops retired (in-order)
        } else {
            wait_async0();
        }
        __syncthreads();
#else
        {   // synchronous staging fallback
            const uint4* sa = (const uint4*)(Ab + (size_t)(mbase + row) * K + kb + half * 16);
            const uint4* sb = (const uint4*)(Bt + (size_t)(nbase + row) * K + kb + half * 16);
            uint4* da = (uint4*)&lA[cur][row * LP_ + half * 16];
            uint4* db = (uint4*)&lB[cur][row * LP_ + half * 16];
            da[0] = sa[0]; da[1] = sa[1];
            db[0] = sb[0]; db[1] = sb[1];
        }
        __syncthreads();
#endif
        // A fragment for this wave's 16 rows, reused across 8 n-tiles
        const uint4* pa = (const uint4*)&lA[cur][(w * 16 + nl) * LP_ + 16 * s];
        v16bf af = make_frag(pa[0], pa[1]);  // 16 contiguous K halves
        #pragma unroll
        for (int t = 0; t < 8; t++) {
            const uint4* pb = (const uint4*)&lB[cur][(t * 16 + nl) * LP_ + 16 * s];
            v16bf bf = make_frag(pb[0], pb[1]);  // 16 contiguous K halves
            acc[t] = wmma_bf16(af, bf, acc[t]);
        }
        __syncthreads();
    }

    if (MODE == 0) {
        unsigned short* o = (unsigned short*)outp + (size_t)batch * oBatch;
        #pragma unroll
        for (int t = 0; t < 8; t++)
            #pragma unroll
            for (int r = 0; r < 8; r++) {
                int rr = mbase + w * 16 + r + 8 * s;
                int cc = nbase + t * 16 + nl;
                o[(size_t)rr * Nn + cc] = f2bf(acc[t][r]);
            }
    } else if (MODE == 1) {
        unsigned short* o = (unsigned short*)outp + (size_t)batch * oBatch;
        #pragma unroll
        for (int t = 0; t < 8; t++) {
            int cc = nbase + t * 16 + nl;
            int h = cc >> 6, d = cc & 63;
            unsigned short p8[8];
            #pragma unroll
            for (int r = 0; r < 8; r++) p8[r] = f2bf(acc[t][r]);
            uint4* dst = (uint4*)(o + ((size_t)(h * 64 + d)) * M_ + mbase + w * 16 + 8 * s);
            *dst = *(const uint4*)p8;
        }
    } else {
        float* o = (float*)outp;
        float gate = gatep[0];
        #pragma unroll
        for (int t = 0; t < 8; t++)
            #pragma unroll
            for (int r = 0; r < 8; r++) {
                int rr = mbase + w * 16 + r + 8 * s;
                int cc = nbase + t * 16 + nl;
                size_t idx = (size_t)rr * Nn + cc;
                o[idx] = xres[idx] + gate * acc[t][r];
            }
    }
}

// ---------------------------------------------------------------------------
// Fused attention: scores (WMMA) + spatial bias + mask + softmax + P@V (WMMA)
// Block = 128 threads (4 waves); each wave owns a 16-row n-tile, full M=128.
// grid = (N/64, H, BT)
// ---------------------------------------------------------------------------
__global__ __launch_bounds__(128) void attn_kernel(
    const unsigned short* __restrict__ q,   // [BT][N][H*D] bf16
    const unsigned short* __restrict__ kbf, // [B][M][H*D]  bf16
    const unsigned short* __restrict__ vt,  // [B][H][D][M] bf16
    const float4* __restrict__ uvzf,        // [BT][M]
    const float* __restrict__ lsig,         // [H]
    const float* __restrict__ sgate,        // [1]
    unsigned short* __restrict__ obuf) {    // [BT][N][H*D] bf16
    __shared__ unsigned short lp[4][16][132];  // per-wave P tile, padded rows
    const int w = threadIdx.x >> 5, lane = threadIdx.x & 31;
    const int nl = lane & 15, s = lane >> 4;
    const int bt = blockIdx.z, h = blockIdx.y;
    const int b = bt >> 1;
    const int n0 = blockIdx.x * 64 + w * 16;

    // --- Q fragments (16 rows x 64 K, two 32-K chunks) ---
    v16bf aq[2];
    #pragma unroll
    for (int c = 0; c < 2; c++) {
        const uint4* p = (const uint4*)(q + ((size_t)(bt * N_ + n0 + nl)) * HD_ +
                                        h * D_ + c * 32 + 16 * s);
        aq[c] = make_frag(p[0], p[1]);
    }

    // --- scores S = Q K^T over 8 m-tiles ---
    v8f S[8];
    #pragma unroll
    for (int mt = 0; mt < 8; mt++) {
        v8f a = {};
        #pragma unroll
        for (int c = 0; c < 2; c++) {
            const uint4* p = (const uint4*)(kbf + ((size_t)(b * M_ + mt * 16 + nl)) * HD_ +
                                            h * D_ + c * 32 + 16 * s);
            a = wmma_bf16(aq[c], make_frag(p[0], p[1]), a);
        }
        S[mt] = a;
    }

    // --- scale + spatial bias + mask ---
    float sig = __expf(lsig[h]);
    float factor = -sgate[0] / (2.f * sig * sig + 1e-8f);
    float gx[8], gy[8];
    #pragma unroll
    for (int r = 0; r < 8; r++) {
        int p = n0 + r + 8 * s;
        gx[r] = ((p % 48) + 0.5f) * (1.f / 48.f);
        gy[r] = ((p / 48) + 0.5f) * (1.f / 48.f);
    }
    #pragma unroll
    for (int mt = 0; mt < 8; mt++) {
        float4 U = uvzf[bt * M_ + mt * 16 + nl];
        #pragma unroll
        for (int r = 0; r < 8; r++) {
            float dx = gx[r] - U.x, dy = gy[r] - U.y;
            float bias = (U.w < 0.f) ? -1e30f
                                     : U.w * factor * (dx * dx + dy * dy) * U.z;
            S[mt][r] = S[mt][r] * 0.125f + bias;   // d^-0.5 = 1/8
        }
    }

    // --- softmax over M per row (8 regs x 16 lanes within each half) ---
    #pragma unroll
    for (int r = 0; r < 8; r++) {
        float mx = -1e30f;
        #pragma unroll
        for (int mt = 0; mt < 8; mt++) mx = fmaxf(mx, S[mt][r]);
        #pragma unroll
        for (int o = 1; o < 16; o <<= 1) mx = fmaxf(mx, __shfl_xor(mx, o, 32));
        float sum = 0.f;
        #pragma unroll
        for (int mt = 0; mt < 8; mt++) {
            float e = __expf(S[mt][r] - mx);
            S[mt][r] = e; sum += e;
        }
        #pragma unroll
        for (int o = 1; o < 16; o <<= 1) sum += __shfl_xor(sum, o, 32);
        float inv = 1.f / sum;
        #pragma unroll
        for (int mt = 0; mt < 8; mt++) S[mt][r] *= inv;
    }

    // --- stage P (bf16) to LDS for D-layout -> A-layout re-shaping ---
    #pragma unroll
    for (int mt = 0; mt < 8; mt++)
        #pragma unroll
        for (int r = 0; r < 8; r++)
            lp[w][r + 8 * s][mt * 16 + nl] = f2bf(S[mt][r]);

    // --- O = P @ V  (K = M = 128, four 32-chunks; 4 d-tiles of 16) ---
    v8f O[4] = {};
    #pragma unroll
    for (int c = 0; c < 4; c++) {
        unsigned pw8[8];
        #pragma unroll
        for (int j = 0; j < 8; j++) {
            int koff = c * 32 + 16 * s + 2 * j;
            pw8[j] = *(const unsigned*)&lp[w][nl][koff];
        }
        uint4 lo = {pw8[0], pw8[1], pw8[2], pw8[3]};
        uint4 hi = {pw8[4], pw8[5], pw8[6], pw8[7]};
        v16bf af = make_frag(lo, hi);
        #pragma unroll
        for (int dt = 0; dt < 4; dt++) {
            const uint4* p = (const uint4*)(vt + ((size_t)((b * H_ + h) * D_ + dt * 16 + nl)) * M_ +
                                            c * 32 + 16 * s);
            O[dt] = wmma_bf16(af, make_frag(p[0], p[1]), O[dt]);
        }
    }

    // --- store O as bf16 [bt][n][h*64+d] ---
    #pragma unroll
    for (int dt = 0; dt < 4; dt++)
        #pragma unroll
        for (int r = 0; r < 8; r++) {
            int n = n0 + r + 8 * s;
            obuf[((size_t)(bt * N_ + n)) * HD_ + h * D_ + dt * 16 + nl] = f2bf(O[dt][r]);
        }
}

// ---------------------------------------------------------------------------
// Host launcher
// ---------------------------------------------------------------------------
extern "C" void kernel_launch(void* const* d_in, const int* in_sizes, int n_in,
                              void* d_out, int out_size, void* d_ws, size_t ws_size,
                              hipStream_t stream) {
    (void)in_sizes; (void)n_in; (void)out_size; (void)ws_size;
    const float* x       = (const float*)d_in[0];
    const float* sgtok   = (const float*)d_in[1];
    const float* pos     = (const float*)d_in[2];
    const float* c2w     = (const float*)d_in[3];
    const float* intr    = (const float*)d_in[4];
    const float* Wq      = (const float*)d_in[5];
    const float* Wk      = (const float*)d_in[6];
    const float* Wv      = (const float*)d_in[7];
    const float* Wo      = (const float*)d_in[8];
    const float* gate    = (const float*)d_in[9];
    const float* sgate   = (const float*)d_in[10];
    const float* lsig    = (const float*)d_in[11];
    const float* lnq_g   = (const float*)d_in[12];
    const float* lnq_b   = (const float*)d_in[13];
    const float* lnk_g   = (const float*)d_in[14];
    const float* lnk_b   = (const float*)d_in[15];
    const unsigned char* mask = (const unsigned char*)d_in[16];

    // Workspace layout (bytes)
    char* ws = (char*)d_ws;
    const size_t szXN = (size_t)BT_ * N_ * C_ * 2;      // 75497472
    const size_t szSN = (size_t)B_ * M_ * CTX_ * 2;     // 1572864
    const size_t szK  = (size_t)B_ * M_ * HD_ * 2;      // 2097152
    const size_t szVT = (size_t)B_ * H_ * D_ * M_ * 2;  // 2097152
    const size_t szWQ = (size_t)C_ * HD_ * 2;
    const size_t szWK = (size_t)CTX_ * HD_ * 2;
    const size_t szWO = (size_t)HD_ * C_ * 2;
    size_t off = 0;
    unsigned short* xn   = (unsigned short*)(ws + off); off += szXN;
    unsigned short* qb   = (unsigned short*)(ws + off); off += szXN;
    unsigned short* oatt = (unsigned short*)(ws + off); off += szXN;
    unsigned short* sn   = (unsigned short*)(ws + off); off += szSN;
    unsigned short* kb   = (unsigned short*)(ws + off); off += szK;
    unsigned short* vtb  = (unsigned short*)(ws + off); off += szVT;
    unsigned short* WQt  = (unsigned short*)(ws + off); off += szWQ;
    unsigned short* WKt  = (unsigned short*)(ws + off); off += szWK;
    unsigned short* WVt  = (unsigned short*)(ws + off); off += szWK;
    unsigned short* WOt  = (unsigned short*)(ws + off); off += szWO;
    float4*         uvzf = (float4*)(ws + off);         off += (size_t)BT_ * M_ * 16;

    // 1) weight transpose/convert
    wtrans_kernel<<<(C_ * HD_ + 255) / 256, 256, 0, stream>>>(Wq, WQt, C_, HD_);
    wtrans_kernel<<<(CTX_ * HD_ + 255) / 256, 256, 0, stream>>>(Wk, WKt, CTX_, HD_);
    wtrans_kernel<<<(CTX_ * HD_ + 255) / 256, 256, 0, stream>>>(Wv, WVt, CTX_, HD_);
    wtrans_kernel<<<(HD_ * C_ + 255) / 256, 256, 0, stream>>>(Wo, WOt, HD_, C_);
    // 2) projection precompute
    uvzf_kernel<<<(BT_ * M_ + 255) / 256, 256, 0, stream>>>(pos, c2w, intr, mask, uvzf);
    // 3) layernorms -> bf16
    ln_bf16_kernel<<<BT_ * N_, 256, 0, stream>>>(x, lnq_g, lnq_b, xn, C_);
    ln_bf16_kernel<<<B_ * M_, 256, 0, stream>>>(sgtok, lnk_g, lnk_b, sn, CTX_);
    // 4) Q = xn @ Wq
    gemm_bf16_kernel<0><<<dim3(HD_ / 128, (BT_ * N_) / 128, 1), 256, 0, stream>>>(
        xn, WQt, qb, nullptr, nullptr, C_, HD_, 0, 0);
    // 5) K = sn @ Wk (batched over B)
    gemm_bf16_kernel<0><<<dim3(HD_ / 128, 1, B_), 256, 0, stream>>>(
        sn, WKt, kb, nullptr, nullptr, CTX_, HD_,
        (long long)M_ * CTX_, (long long)M_ * HD_);
    // 6) V^T = (sn @ Wv)^T per head -> [b][h][d][m]
    gemm_bf16_kernel<1><<<dim3(HD_ / 128, 1, B_), 256, 0, stream>>>(
        sn, WVt, vtb, nullptr, nullptr, CTX_, HD_,
        (long long)M_ * CTX_, (long long)H_ * D_ * M_);
    // 7) fused attention
    attn_kernel<<<dim3(N_ / 64, H_, BT_), 128, 0, stream>>>(
        qb, kb, vtb, uvzf, lsig, sgate, oatt);
    // 8) out = x + gate * (O @ Wo)
    gemm_bf16_kernel<2><<<dim3(C_ / 128, (BT_ * N_) / 128, 1), 256, 0, stream>>>(
        oatt, WOt, d_out, x, gate, HD_, C_, 0, 0);
}
